// CopyMemoryModel_25443386261682
// MI455X (gfx1250) — matrix operand order repaired and checked by reading, not compiled, not run on hardware
//
#include <hip/hip_runtime.h>
#include <cmath>

typedef __bf16 v16bf __attribute__((ext_vector_type(16)));
typedef __bf16 v8bf  __attribute__((ext_vector_type(8)));
typedef float  v8f   __attribute__((ext_vector_type(8)));

namespace {
constexpr int Bn = 64, Sn = 128, DINn = 64, Dn = 1024, Hn = 512, DOUTn = 64;
constexpr int Mtot = Bn * Sn;          // 8192 independent rows
constexpr int TM = 32;                 // rows per workgroup (2 WMMA M-tiles)
constexpr int NTHREADS = 256;          // 8 wave32s
constexpr int CUR_LD = 1025;           // fp32 row stride (pad: avoid bank conflicts)
constexpr int AB_LD  = 1048;           // bf16 row stride: 524 dwords == 12 mod 64 banks, 16B aligned

constexpr size_t SZ_WIN  = (size_t)DINn * Dn * 2;
constexpr size_t SZ_FULL = (size_t)Dn * Dn * 2;
constexpr size_t SZ_HALF = (size_t)Hn * Hn * 2;
constexpr size_t SZ_WO   = (size_t)Dn * DOUTn * 2;
constexpr size_t OFF_WIN  = 0;
constexpr size_t OFF_FULL = OFF_WIN + SZ_WIN;            // 9 full 1024x1024 mats
constexpr size_t OFF_HALF = OFF_FULL + 9 * SZ_FULL;      // F0..2, G0..2
constexpr size_t OFF_WO   = OFF_HALF + 6 * SZ_HALF;      // 1024x64
// total ws use: ~22.3 MB (weights bf16, L2-resident on 192MB L2)

constexpr size_t SMEM_BYTES = (size_t)TM * CUR_LD * 4 + (size_t)TM * AB_LD * 2;
} // namespace

// ---------------------------------------------------------------------------
// Weight prep: convert fp32 source matrices into bf16 "B-fragment" layout:
// block(ntile,kt) of 32x16, stored so lane L's 16 bf16 values are contiguous:
//   n = ntile*16 + (L&15);  k = kt*32 + (L>>4)*8 + (e&7) + (e>>3)*16  (e = 0..15)
// matching V_WMMA_*_16X16X32 B-operand VGPR packing (K pairs per dword).
// mode 0: W[k][n]   mode 1: W[n][k] (transpose)   mode 2: R[k][n]+R[n][k]
// ---------------------------------------------------------------------------
__global__ void enigma_prep_bfrag(const float* __restrict__ src, __bf16* __restrict__ dst,
                                  int Ktot, int Ntot, int mode, int srcLD) {
  const int total = Ktot * Ntot;
  const int kt32 = Ktot >> 5;
  for (int tid = blockIdx.x * blockDim.x + threadIdx.x; tid < total;
       tid += gridDim.x * blockDim.x) {
    int e     = tid & 15;
    int lane  = (tid >> 4) & 31;
    int blk   = tid >> 9;
    int kt    = blk % kt32;
    int ntile = blk / kt32;
    int n = ntile * 16 + (lane & 15);
    int k = kt * 32 + (lane >> 4) * 8 + (e & 7) + (e >> 3) * 16;
    float v;
    if (mode == 0)      v = src[(size_t)k * srcLD + n];
    else if (mode == 1) v = src[(size_t)n * srcLD + k];
    else                v = src[(size_t)k * Ntot + n] + src[(size_t)n * Ntot + k];
    dst[tid] = (__bf16)v;
  }
}

// ---------------------------------------------------------------------------
__device__ __forceinline__ v8f wmma_bf16(v16bf a, v16bf b, v8f c) {
  return __builtin_amdgcn_wmma_f32_16x16x32_bf16(false, a, false, b, (short)0, c,
                                                 false, false);
}

// A fragment from row-major bf16 LDS tile, per ISA 16-bit A 16x32 layout:
// lanes 0-15: M=lane,  K = kbase+{0..7} and kbase+16+{0..7}
// lanes16-31: M=lane-16, same chunks shifted by +8.
__device__ __forceinline__ v16bf load_a_frag(const __bf16* Abuf, int mt, int kt, int lane) {
  const int half = lane >> 4;
  const int m = mt * 16 + (lane & 15);
  const __bf16* p = Abuf + m * AB_LD + kt * 32 + half * 8;
  v8bf c0 = *(const v8bf*)(p);        // ds_load_b128
  v8bf c1 = *(const v8bf*)(p + 16);   // ds_load_b128
  v16bf a;
#pragma unroll
  for (int i = 0; i < 8; ++i) { a[i] = c0[i]; a[i + 8] = c1[i]; }
  return a;
}

// B fragment: 32 contiguous bytes per lane from the prepped layout.
__device__ __forceinline__ v16bf load_b_frag(const __bf16* Wf, int kt, int nt,
                                             int ktiles, int lane) {
  const __bf16* p = Wf + (((size_t)nt * ktiles + kt) * 32 + lane) * 16;
  return *(const v16bf*)(p);          // 2x global_load_b128
}

// ---------------------------------------------------------------------------
__global__ __launch_bounds__(NTHREADS, 1) void enigma_fused_kernel(
    const float* __restrict__ x, const float* __restrict__ bi,
    const float* __restrict__ bo, const __bf16* __restrict__ ws,
    float* __restrict__ out) {
  extern __shared__ char smem_raw[];
  float*  cur  = (float*)smem_raw;                                   // [TM][CUR_LD] fp32
  __bf16* Abuf = (__bf16*)(smem_raw + (size_t)TM * CUR_LD * 4);      // [TM][AB_LD] bf16

  const int tid  = threadIdx.x;
  const int lane = tid & 31;
  const int wave = tid >> 5;
  const int row0 = blockIdx.x * TM;

  auto pos_of = [&](int m, int rot) -> int {
    int t = (row0 + m) & (Sn - 1);                 // t = row % S
    return (t >> (10 * rot)) & (Dn - 1);           // (t / D^rot) % D
  };

  // pack cur (optionally rolled left by pos) -> Abuf bf16
  auto pack_full = [&](int rot) {
    for (int idx = tid; idx < TM * Dn; idx += NTHREADS) {
      int m = idx >> 10, k = idx & (Dn - 1);
      int ksrc = (rot >= 0) ? ((k + pos_of(m, rot)) & (Dn - 1)) : k;
      Abuf[m * AB_LD + k] = (__bf16)cur[m * CUR_LD + ksrc];
    }
  };
  auto pack_half = [&](int srcOff) {
    for (int idx = tid; idx < TM * Hn; idx += NTHREADS) {
      int m = idx >> 9, k = idx & (Hn - 1);
      Abuf[m * AB_LD + k] = (__bf16)cur[m * CUR_LD + srcOff + k];
    }
  };

  // N=1024 GEMM: per wave 8 N-tiles x 2 M-tiles held in accumulators.
  // wmode: 0 plain store, 1 rolled-output store (rotor), 2 add input bias.
  auto gemm_full = [&](const __bf16* Wf, int ktiles, int wmode, int rot) {
    v8f acc[2][8] = {};
    for (int kt = 0; kt < ktiles; ++kt) {
      v16bf a0 = load_a_frag(Abuf, 0, kt, lane);
      v16bf a1 = load_a_frag(Abuf, 1, kt, lane);
#pragma unroll
      for (int i = 0; i < 8; ++i) {
        v16bf b = load_b_frag(Wf, kt, wave * 8 + i, ktiles, lane);
        acc[0][i] = wmma_bf16(a0, b, acc[0][i]);
        acc[1][i] = wmma_bf16(a1, b, acc[1][i]);
      }
    }
    const int half = lane >> 4;
#pragma unroll
    for (int i = 0; i < 8; ++i) {
      int ncol = (wave * 8 + i) * 16 + (lane & 15);
#pragma unroll
      for (int mt = 0; mt < 2; ++mt) {
#pragma unroll
        for (int r = 0; r < 8; ++r) {              // C/D layout: M = 8*half + r
          int m = mt * 16 + half * 8 + r;
          float v = acc[mt][i][r];
          int col = ncol;
          if (wmode == 1)      col = (ncol + pos_of(m, rot)) & (Dn - 1);
          else if (wmode == 2) v += bi[ncol];
          cur[m * CUR_LD + col] = v;
        }
      }
    }
  };

  // 512x512 reversible half-block GEMM: dst_half += tanh(src_half @ W)
  auto gemm_half = [&](const __bf16* Wf, int dstOff) {
    v8f acc[2][4] = {};
    for (int kt = 0; kt < Hn / 32; ++kt) {
      v16bf a0 = load_a_frag(Abuf, 0, kt, lane);
      v16bf a1 = load_a_frag(Abuf, 1, kt, lane);
#pragma unroll
      for (int i = 0; i < 4; ++i) {
        v16bf b = load_b_frag(Wf, kt, wave * 4 + i, Hn / 32, lane);
        acc[0][i] = wmma_bf16(a0, b, acc[0][i]);
        acc[1][i] = wmma_bf16(a1, b, acc[1][i]);
      }
    }
    const int half = lane >> 4;
#pragma unroll
    for (int i = 0; i < 4; ++i) {
      int ncol = dstOff + (wave * 4 + i) * 16 + (lane & 15);
#pragma unroll
      for (int mt = 0; mt < 2; ++mt) {
#pragma unroll
        for (int r = 0; r < 8; ++r) {
          int m = mt * 16 + half * 8 + r;
          float* p = &cur[m * CUR_LD + ncol];
          *p = *p + tanhf(acc[mt][i][r]);
        }
      }
    }
  };

  auto Wfull = [&](int i) { return ws + (OFF_FULL + i * SZ_FULL) / 2; };
  auto Whalf = [&](int i) { return ws + (OFF_HALF + i * SZ_HALF) / 2; };

  // ---- stage 0: h = x @ Wi^T + bi  (K=64) ----
  for (int idx = tid; idx < TM * DINn; idx += NTHREADS) {
    int m = idx >> 6, k = idx & (DINn - 1);
    Abuf[m * AB_LD + k] = (__bf16)x[(size_t)(row0 + m) * DINn + k];
  }
  __syncthreads();
  gemm_full(ws + OFF_WIN / 2, DINn / 32, 2, 0);
  __syncthreads();

  // ---- plugboard forward: h @ P^T ----
  pack_full(-1); __syncthreads();
  gemm_full(Wfull(0), Dn / 32, 0, 0); __syncthreads();

  // ---- rotors forward: roll(-pos) @ rotW[i] then roll(+pos) ----
  for (int r = 0; r < 3; ++r) {
    pack_full(r); __syncthreads();
    gemm_full(Wfull(1 + r), Dn / 32, 1, r); __syncthreads();
  }

  // ---- reversible blocks forward (i = 0,1,2) ----
  for (int i = 0; i < 3; ++i) {
    pack_half(Hn); __syncthreads();           // y1 = h1 + tanh(h2 @ F[i])
    gemm_half(Whalf(i), 0); __syncthreads();
    pack_half(0); __syncthreads();            // y2 = h2 + tanh(y1 @ G[i])
    gemm_half(Whalf(3 + i), Hn); __syncthreads();
  }

  // ---- reflector: h @ (R + R^T) ----
  pack_full(-1); __syncthreads();
  gemm_full(Wfull(4), Dn / 32, 0, 0); __syncthreads();

  // ---- reversible blocks, reversed order (same forward blocks) ----
  for (int i = 2; i >= 0; --i) {
    pack_half(Hn); __syncthreads();
    gemm_half(Whalf(i), 0); __syncthreads();
    pack_half(0); __syncthreads();
    gemm_half(Whalf(3 + i), Hn); __syncthreads();
  }

  // ---- inverse rotors i = 2,1,0 with rotW[i]^T ----
  for (int i = 2; i >= 0; --i) {
    pack_full(i); __syncthreads();
    gemm_full(Wfull(5 + (2 - i)), Dn / 32, 1, i); __syncthreads();
  }

  // ---- plugboard return: h @ P ----
  pack_full(-1); __syncthreads();
  gemm_full(Wfull(8), Dn / 32, 0, 0); __syncthreads();

  // ---- output projection: out = h @ Wo^T + bo  (N=64: one tile per wave) ----
  pack_full(-1); __syncthreads();
  {
    const __bf16* Wwo = ws + OFF_WO / 2;
    v8f acc = {};
    const int nt = wave & 3, mt = wave >> 2;
    for (int kt = 0; kt < Dn / 32; ++kt) {
      v16bf a = load_a_frag(Abuf, mt, kt, lane);
      v16bf b = load_b_frag(Wwo, kt, nt, Dn / 32, lane);
      acc = wmma_bf16(a, b, acc);
    }
    const int half = lane >> 4;
    const int ncol = nt * 16 + (lane & 15);
    const float bias = bo[ncol];
#pragma unroll
    for (int r = 0; r < 8; ++r) {
      int m = mt * 16 + half * 8 + r;
      out[(size_t)(row0 + m) * DOUTn + ncol] = acc[r] + bias;
    }
  }
}

// ---------------------------------------------------------------------------
extern "C" void kernel_launch(void* const* d_in, const int* in_sizes, int n_in,
                              void* d_out, int out_size, void* d_ws, size_t ws_size,
                              hipStream_t stream) {
  (void)in_sizes; (void)n_in; (void)out_size; (void)ws_size;
  const float* x    = (const float*)d_in[0];
  const float* Wi   = (const float*)d_in[1];
  const float* bi   = (const float*)d_in[2];
  const float* P    = (const float*)d_in[3];
  const float* rotW = (const float*)d_in[4];
  const float* F    = (const float*)d_in[5];
  const float* G    = (const float*)d_in[6];
  const float* R    = (const float*)d_in[7];
  const float* Wo   = (const float*)d_in[8];
  const float* bo   = (const float*)d_in[9];
  float* out  = (float*)d_out;
  __bf16* wsb = (__bf16*)d_ws;

  auto prep = [&](const float* src, size_t dstOffBytes, int K, int N, int mode, int ld) {
    int total = K * N;
    int grid = (total + 255) / 256;
    if (grid > 4096) grid = 4096;
    enigma_prep_bfrag<<<grid, 256, 0, stream>>>(src, wsb + dstOffBytes / 2, K, N, mode, ld);
  };

  // input proj: Win_eff[k][n] = Wi[n][k]
  prep(Wi, OFF_WIN, DINn, Dn, 1, DINn);
  // full-D chain mats: [P^T, rot0, rot1, rot2, R+R^T, rot2^T, rot1^T, rot0^T, P]
  prep(P, OFF_FULL + 0 * SZ_FULL, Dn, Dn, 1, Dn);
  for (int i = 0; i < 3; ++i)
    prep(rotW + (size_t)i * Dn * Dn, OFF_FULL + (1 + i) * SZ_FULL, Dn, Dn, 0, Dn);
  prep(R, OFF_FULL + 4 * SZ_FULL, Dn, Dn, 2, Dn);
  for (int i = 0; i < 3; ++i)  // 5:rot2^T 6:rot1^T 7:rot0^T
    prep(rotW + (size_t)(2 - i) * Dn * Dn, OFF_FULL + (5 + i) * SZ_FULL, Dn, Dn, 1, Dn);
  prep(P, OFF_FULL + 8 * SZ_FULL, Dn, Dn, 0, Dn);
  // reversible-block mats
  for (int i = 0; i < 3; ++i) {
    prep(F + (size_t)i * Hn * Hn, OFF_HALF + i * SZ_HALF, Hn, Hn, 0, Hn);
    prep(G + (size_t)i * Hn * Hn, OFF_HALF + (3 + i) * SZ_HALF, Hn, Hn, 0, Hn);
  }
  // output proj: Wo_eff[k][n] = Wo[n][k]
  prep(Wo, OFF_WO, Dn, DOUTn, 1, Dn);

  enigma_fused_kernel<<<Mtot / TM, NTHREADS, SMEM_BYTES, stream>>>(x, bi, bo, wsb, out);
}
